// OGRENet_50422916055679
// MI455X (gfx1250) — matrix operand order repaired and checked by reading, not compiled, not run on hardware
//
#include <hip/hip_runtime.h>
#include <stdint.h>

// ---------------- CDNA5 WMMA types ----------------
typedef __attribute__((ext_vector_type(16))) __bf16 v16bf;
typedef __attribute__((ext_vector_type(8)))  float  v8f;

union FragAB {
    uint4 q[2];   // 32 bytes = 16 bf16 = 8 VGPRs
    v16bf v;
};

#define LDW      1040   // LDS row stride in bf16 elems (16B aligned, bank-swizzled)
#define TILE_E   64
#define NTHREADS 256
#define NWAVES   8

__device__ __forceinline__ unsigned short f2bf(float f) {
    unsigned int u = __float_as_uint(f);
    unsigned int r = u + 0x7FFFu + ((u >> 16) & 1u);   // round-to-nearest-even
    return (unsigned short)(r >> 16);
}
__device__ __forceinline__ float bf2f(unsigned short h) {
    return __uint_as_float(((unsigned int)h) << 16);
}

// ---------------------------------------------------------------------------
// Tile GEMM:  D[64 x Nout] = relu?( A[64 x K] * W^T + bias )
//   A in LDS (bf16, row stride LDW), W bf16 global [Nout][K] (K mult of 32),
//   D written back to LDS as bf16.
//   Each wave owns ONE 16-wide N-tile and ALL FOUR 16-row M-tiles: every B
//   fragment is loaded exactly once per workgroup and feeds 4 independent
//   WMMAs (latency hiding + 4x less L2 weight traffic).
// ---------------------------------------------------------------------------
__device__ __forceinline__ void gemm_tile(
    const unsigned short* __restrict__ aBuf,
    unsigned short* __restrict__ dBuf,
    const unsigned short* __restrict__ Wbf,
    const float* __restrict__ bias,
    int K, int Nout, bool relu)
{
    const int wave = threadIdx.x >> 5;
    const int lane = threadIdx.x & 31;
    const int nlo  = lane & 15;
    const int half = lane >> 4;
    const int nT   = Nout >> 4;                     // N-tiles

    for (int t = wave; t < nT; t += NWAVES) {
        const int n0 = t << 4;

        const float b = bias[n0 + nlo];
        v8f acc0 = {b, b, b, b, b, b, b, b};
        v8f acc1 = acc0, acc2 = acc0, acc3 = acc0;

        // A: lane<16 -> K {0..7, 16..23};  lane>=16 -> K {8..15, 24..31}
        const unsigned short* arow = aBuf + nlo * LDW + half * 8;
        // B: lane<16 -> K {0..7, 16..23};  lane>=16 -> K {8..15, 24..31}   (row = n0+nlo)
        const unsigned short* wrow = Wbf + (size_t)(n0 + nlo) * K + half * 8;

        #pragma unroll 2
        for (int k0 = 0; k0 < K; k0 += 32) {
            FragAB w;
            w.q[0] = *(const uint4*)(wrow + k0);
            w.q[1] = *(const uint4*)(wrow + k0 + 16);

            FragAB a0, a1, a2, a3;
            a0.q[0] = *(const uint4*)(arow + k0);
            a0.q[1] = *(const uint4*)(arow + k0 + 16);
            a1.q[0] = *(const uint4*)(arow + 16 * LDW + k0);
            a1.q[1] = *(const uint4*)(arow + 16 * LDW + k0 + 16);
            a2.q[0] = *(const uint4*)(arow + 32 * LDW + k0);
            a2.q[1] = *(const uint4*)(arow + 32 * LDW + k0 + 16);
            a3.q[0] = *(const uint4*)(arow + 48 * LDW + k0);
            a3.q[1] = *(const uint4*)(arow + 48 * LDW + k0 + 16);

            acc0 = __builtin_amdgcn_wmma_f32_16x16x32_bf16(
                false, a0.v, false, w.v, (short)0, acc0, false, false);
            acc1 = __builtin_amdgcn_wmma_f32_16x16x32_bf16(
                false, a1.v, false, w.v, (short)0, acc1, false, false);
            acc2 = __builtin_amdgcn_wmma_f32_16x16x32_bf16(
                false, a2.v, false, w.v, (short)0, acc2, false, false);
            acc3 = __builtin_amdgcn_wmma_f32_16x16x32_bf16(
                false, a3.v, false, w.v, (short)0, acc3, false, false);
        }

        // D layout: VGPR r -> (M = mtile*16 + r + half*8, N = n0 + nlo)
        unsigned short* dcol = dBuf + n0 + nlo;
        #pragma unroll 8
        for (int r = 0; r < 8; ++r) {
            const int mr = r + half * 8;
            float v0 = acc0[r], v1 = acc1[r], v2 = acc2[r], v3 = acc3[r];
            if (relu) {
                v0 = fmaxf(v0, 0.f); v1 = fmaxf(v1, 0.f);
                v2 = fmaxf(v2, 0.f); v3 = fmaxf(v3, 0.f);
            }
            dcol[(mr     ) * LDW] = f2bf(v0);
            dcol[(mr + 16) * LDW] = f2bf(v1);
            dcol[(mr + 32) * LDW] = f2bf(v2);
            dcol[(mr + 48) * LDW] = f2bf(v3);
        }
    }
}

// ---------------------------------------------------------------------------
// u_r = u @ w_sel^T + b_sel        [16 x 4096] x [4096 x 256] -> [16 x 256]
// ---------------------------------------------------------------------------
__global__ void u_reduce_kernel(const float* __restrict__ u,
                                const float* __restrict__ w_sel,
                                const float* __restrict__ b_sel,
                                float* __restrict__ u_r)
{
    const int j = threadIdx.x;           // 0..255 output feature
    const int b = blockIdx.x;            // graph
    float acc = b_sel[j];
    const float* urow = u     + (size_t)b * 4096;
    const float* wrow = w_sel + (size_t)j * 4096;
    for (int k = 0; k < 4096; k += 4) {
        float4 a = *(const float4*)(urow + k);
        float4 w = *(const float4*)(wrow + k);
        acc += a.x * w.x + a.y * w.y + a.z * w.z + a.w * w.w;
    }
    u_r[b * 256 + j] = acc;
}

// fp32 [N][K] -> bf16 [N][Kp] (tail-padded with zeros)
__global__ void pad_bf16_kernel(const float* __restrict__ src,
                                unsigned short* __restrict__ dst,
                                int Nrows, int K, int Kp)
{
    const int total = Nrows * Kp;
    for (int idx = blockIdx.x * blockDim.x + threadIdx.x; idx < total;
         idx += gridDim.x * blockDim.x) {
        const int n = idx / Kp, k = idx % Kp;
        dst[idx] = (k < K) ? f2bf(src[(size_t)n * K + k]) : (unsigned short)0;
    }
}

// ---------------------------------------------------------------------------
// Fused edge pipeline: gather -> edge MLP (5 layers) -> node MLP1 (2 layers)
//                      -> scatter-mean accumulation.  64 edges per workgroup.
// ---------------------------------------------------------------------------
__global__ __launch_bounds__(NTHREADS)
void edge_fused_kernel(
    const float* __restrict__ x,
    const int*   __restrict__ edge_index,   // [2,E]
    const float* __restrict__ edge_attr,    // [E]
    const int*   __restrict__ batch,        // [N]
    const float* __restrict__ u_r,          // [16,256]
    const unsigned short* __restrict__ ew0b, const float* __restrict__ eb0,
    const unsigned short* __restrict__ ew1b, const float* __restrict__ eb1,
    const unsigned short* __restrict__ ew2b, const float* __restrict__ eb2,
    const unsigned short* __restrict__ ew3b, const float* __restrict__ eb3,
    const unsigned short* __restrict__ ew4b, const float* __restrict__ eb4,
    const unsigned short* __restrict__ nw0b, const float* __restrict__ nb0,
    const unsigned short* __restrict__ nw1b, const float* __restrict__ nb1,
    float* __restrict__ seg_sum,            // [N,512]
    float* __restrict__ cnt,                // [N]
    int E)
{
    extern __shared__ unsigned short smem[];
    unsigned short* buf0 = smem;
    unsigned short* buf1 = smem + TILE_E * LDW;
    const int base = blockIdx.x * TILE_E;
    const int tid  = threadIdx.x;

    // ---- gather e_in tile [64 x 288] = [x[row] | x[col] | ea | u_r[b] | 0pad]
    {
        const int r = tid >> 2;
        const int e = base + r;
        const bool valid = (e < E);
        int src = 0, dst = 0, eb = 0;
        if (valid) {
            src = edge_index[e];
            dst = edge_index[E + e];
            eb  = batch[src];
        }
        const int c0 = (tid & 3) * 72;
        for (int c = c0; c < c0 + 72; ++c) {
            float v = 0.f;
            if (valid) {
                if (c < 9)        v = x[src * 9 + c];
                else if (c < 18)  v = x[dst * 9 + (c - 9)];
                else if (c == 18) v = edge_attr[e];
                else if (c < 275) v = u_r[eb * 256 + (c - 19)];
            }
            buf0[r * LDW + c] = f2bf(v);
        }
    }
    __syncthreads();

    gemm_tile(buf0, buf1, ew0b, eb0,  288, 1024, true);  __syncthreads();
    gemm_tile(buf1, buf0, ew1b, eb1, 1024, 1024, true);  __syncthreads();
    gemm_tile(buf0, buf1, ew2b, eb2, 1024, 1024, true);  __syncthreads();
    gemm_tile(buf1, buf0, ew3b, eb3, 1024, 1024, true);  __syncthreads();
    gemm_tile(buf0, buf1, ew4b, eb4, 1024,  512, false); __syncthreads();
    // buf1 = edge_out [64 x 512]

    // ---- build n_in [64 x 544] = [x[col](9) | edge_out(512) | 0pad]
    {
        const int r = tid >> 2;
        const int e = base + r;
        const bool valid = (e < E);
        const int dst = valid ? edge_index[E + e] : 0;
        const int c0 = (tid & 3) * 136;
        for (int c = c0; c < c0 + 136; ++c) {
            unsigned short v;
            if (c < 9)        v = valid ? f2bf(x[dst * 9 + c]) : (unsigned short)0;
            else if (c < 521) v = buf1[r * LDW + (c - 9)];
            else              v = 0;
            buf0[r * LDW + c] = v;
        }
    }
    __syncthreads();
    gemm_tile(buf0, buf1, nw0b, nb0, 544, 512, true); __syncthreads();
    gemm_tile(buf1, buf0, nw1b, nb1, 512, 512, true); __syncthreads();
    // buf0 = h [64 x 512]

    // ---- scatter-mean accumulation onto row ----
    for (int idx = tid; idx < TILE_E * 512; idx += NTHREADS) {
        const int r = idx >> 9;
        const int c = idx & 511;
        const int e = base + r;
        if (e < E) {
            const int src = edge_index[e];
            atomicAdd(&seg_sum[(size_t)src * 512 + c], bf2f(buf0[r * LDW + c]));
        }
    }
    if (tid < TILE_E) {
        const int e = base + tid;
        if (e < E) atomicAdd(&cnt[edge_index[e]], 1.0f);
    }
}

// ---------------------------------------------------------------------------
// Final per-node MLP: m_in = [x | agg | u_r[batch]] (777 -> 800 padded)
//   h2 = relu(m_in @ mw0^T + mb0);  out = h2 @ mw1^T + mb1
// ---------------------------------------------------------------------------
__global__ __launch_bounds__(NTHREADS)
void node_final_kernel(
    const float* __restrict__ x,
    const int*   __restrict__ batch,
    const float* __restrict__ u_r,
    const float* __restrict__ seg_sum,
    const float* __restrict__ cnt,
    const unsigned short* __restrict__ mw0b, const float* __restrict__ mb0,
    const float* __restrict__ mw1, const float* __restrict__ mb1,
    float* __restrict__ out, int N)
{
    extern __shared__ unsigned short smem[];
    unsigned short* buf0 = smem;
    unsigned short* buf1 = smem + TILE_E * LDW;
    const int base = blockIdx.x * TILE_E;
    const int tid  = threadIdx.x;

    {
        const int r = tid >> 2;
        const int i = base + r;
        const bool valid = (i < N);
        float inv = 1.f;
        int gb = 0;
        if (valid) {
            inv = 1.f / fmaxf(cnt[i], 1.f);
            gb  = batch[i];
        }
        const int c0 = (tid & 3) * 200;
        for (int c = c0; c < c0 + 200; ++c) {
            float v = 0.f;
            if (valid) {
                if (c < 9)        v = x[i * 9 + c];
                else if (c < 521) v = seg_sum[(size_t)i * 512 + (c - 9)] * inv;
                else if (c < 777) v = u_r[gb * 256 + (c - 521)];
            }
            buf0[r * LDW + c] = f2bf(v);
        }
    }
    __syncthreads();
    gemm_tile(buf0, buf1, mw0b, mb0, 800, 512, true);
    __syncthreads();

    if (tid < TILE_E) {
        const int i = base + tid;
        if (i < N) {
            float acc = mb1[0];
            for (int c = 0; c < 512; ++c)
                acc += bf2f(buf1[tid * LDW + c]) * mw1[c];
            out[i] = acc;
        }
    }
}

// ---------------------------------------------------------------------------
extern "C" void kernel_launch(void* const* d_in, const int* in_sizes, int n_in,
                              void* d_out, int out_size, void* d_ws, size_t ws_size,
                              hipStream_t stream) {
    // inputs in setup_inputs() order
    const float* x          = (const float*)d_in[0];
    const int*   edge_index = (const int*)  d_in[1];
    const float* edge_attr  = (const float*)d_in[2];
    const float* u          = (const float*)d_in[3];
    const int*   batch      = (const int*)  d_in[4];
    const float* w_sel = (const float*)d_in[5];  const float* b_sel = (const float*)d_in[6];
    const float* ew0 = (const float*)d_in[7];    const float* eb0 = (const float*)d_in[8];
    const float* ew1 = (const float*)d_in[9];    const float* eb1 = (const float*)d_in[10];
    const float* ew2 = (const float*)d_in[11];   const float* eb2 = (const float*)d_in[12];
    const float* ew3 = (const float*)d_in[13];   const float* eb3 = (const float*)d_in[14];
    const float* ew4 = (const float*)d_in[15];   const float* eb4 = (const float*)d_in[16];
    const float* nw0 = (const float*)d_in[17];   const float* nb0 = (const float*)d_in[18];
    const float* nw1 = (const float*)d_in[19];   const float* nb1 = (const float*)d_in[20];
    const float* mw0 = (const float*)d_in[21];   const float* mb0 = (const float*)d_in[22];
    const float* mw1 = (const float*)d_in[23];   const float* mb1 = (const float*)d_in[24];

    const int N = in_sizes[0] / 9;
    const int E = in_sizes[2];            // edge_attr is [E,1]
    const int B = in_sizes[3] / 4096;

    // ---- workspace layout (256B aligned) ----
    char* ws = (char*)d_ws;
    size_t off = 0;
    auto alloc = [&](size_t bytes) { char* p = ws + off; off = (off + bytes + 255) & ~(size_t)255; return p; };
    float*          ws_ur   = (float*)alloc((size_t)B * 256 * 4);
    float*          ws_seg  = (float*)alloc((size_t)N * 512 * 4);
    float*          ws_cnt  = (float*)alloc((size_t)N * 4);
    unsigned short* ws_ew0  = (unsigned short*)alloc((size_t)1024 * 288 * 2);
    unsigned short* ws_ew1  = (unsigned short*)alloc((size_t)1024 * 1024 * 2);
    unsigned short* ws_ew2  = (unsigned short*)alloc((size_t)1024 * 1024 * 2);
    unsigned short* ws_ew3  = (unsigned short*)alloc((size_t)1024 * 1024 * 2);
    unsigned short* ws_ew4  = (unsigned short*)alloc((size_t)512 * 1024 * 2);
    unsigned short* ws_nw0  = (unsigned short*)alloc((size_t)512 * 544 * 2);
    unsigned short* ws_nw1  = (unsigned short*)alloc((size_t)512 * 512 * 2);
    unsigned short* ws_mw0  = (unsigned short*)alloc((size_t)512 * 800 * 2);
    (void)ws_size; (void)n_in; (void)out_size;

    // ---- u reduction + weight bf16 prep ----
    u_reduce_kernel<<<B, 256, 0, stream>>>(u, w_sel, b_sel, ws_ur);
    auto pad = [&](const float* src, unsigned short* dst, int Nr, int K, int Kp) {
        int total = Nr * Kp;
        int blocks = (total + 1023) / 1024;
        pad_bf16_kernel<<<blocks, 256, 0, stream>>>(src, dst, Nr, K, Kp);
    };
    pad(ew0, ws_ew0, 1024, 275, 288);
    pad(ew1, ws_ew1, 1024, 1024, 1024);
    pad(ew2, ws_ew2, 1024, 1024, 1024);
    pad(ew3, ws_ew3, 1024, 1024, 1024);
    pad(ew4, ws_ew4, 512, 1024, 1024);
    pad(nw0, ws_nw0, 512, 521, 544);
    pad(nw1, ws_nw1, 512, 512, 512);
    pad(mw0, ws_mw0, 512, 777, 800);

    hipMemsetAsync(ws_seg, 0, (size_t)N * 512 * 4, stream);
    hipMemsetAsync(ws_cnt, 0, (size_t)N * 4, stream);

    const size_t smemBytes = (size_t)2 * TILE_E * LDW * sizeof(unsigned short); // ~260 KB
    hipFuncSetAttribute((const void*)edge_fused_kernel,
                        hipFuncAttributeMaxDynamicSharedMemorySize, (int)smemBytes);
    hipFuncSetAttribute((const void*)node_final_kernel,
                        hipFuncAttributeMaxDynamicSharedMemorySize, (int)smemBytes);

    const int edgeBlocks = (E + TILE_E - 1) / TILE_E;
    edge_fused_kernel<<<edgeBlocks, NTHREADS, smemBytes, stream>>>(
        x, edge_index, edge_attr, batch, ws_ur,
        ws_ew0, eb0, ws_ew1, eb1, ws_ew2, eb2, ws_ew3, eb3, ws_ew4, eb4,
        ws_nw0, nb0, ws_nw1, nb1,
        ws_seg, ws_cnt, E);

    const int nodeBlocks = (N + TILE_E - 1) / TILE_E;
    node_final_kernel<<<nodeBlocks, NTHREADS, smemBytes, stream>>>(
        x, batch, ws_ur, ws_seg, ws_cnt,
        ws_mw0, mb0, mw1, mb1, (float*)d_out, N);
}
